// MultiHeadAttention_3693671874746
// MI455X (gfx1250) — matrix operand order repaired and checked by reading, not compiled, not run on hardware
//
#include <hip/hip_runtime.h>
#include <hip/hip_bf16.h>

// ---------------------------------------------------------------------------
// Complex multi-head attention for MI455X (gfx1250, wave32, WMMA).
// B=4, Nq=Nk=1024, R=256, NH=8, DK=DV=64.
// Pipeline: [proj Q/K/V -> bf16] -> [fused flash attention, complex,
//            double-buffered global_load_async_to_lds staging] -> [out proj].
// All matrix math on v_wmma_f32_16x16x32_bf16 (fp32 accumulate).
// ---------------------------------------------------------------------------

#define NB   4
#define NSEQ 1024
#define RDIM 256
#define NHH  8
#define DHD  64
#define CPROJ (NHH * DHD)   // 512

typedef __attribute__((ext_vector_type(16))) __bf16 v16bf;
typedef __attribute__((ext_vector_type(8)))  float  v8f;

static __device__ __forceinline__ v8f wmma_bf16(v16bf a, v16bf b, v8f c) {
  // (neg_a, A, neg_b, B, c_mod, C, reuse_a, reuse_b)
  return __builtin_amdgcn_wmma_f32_16x16x32_bf16(false, a, false, b, (short)0, c,
                                                 false, false);
}

static __device__ __forceinline__ v8f v8f_zero() {
  v8f z = {0.f, 0.f, 0.f, 0.f, 0.f, 0.f, 0.f, 0.f};
  return z;
}

// Async 16-byte global->LDS copy (CDNA5, ASYNCcnt-tracked).
// GVS addressing: mem = SGPR64 base + i32 VGPR offset. VDST = LDS byte addr.
static __device__ __forceinline__ void async_copy16(unsigned ldsoff,
                                                    unsigned goff,
                                                    const void* base) {
  asm volatile("global_load_async_to_lds_b128 %0, %1, %2"
               :
               : "v"(ldsoff), "v"(goff),
                 "s"((unsigned long long)(uintptr_t)base)
               : "memory");
}

static __device__ __forceinline__ void wait_asynccnt0() {
  asm volatile("s_wait_asynccnt 0x0" ::: "memory");
}
// Double-buffer wait: allow the 8 just-issued loads to remain in flight;
// ASYNCcnt completes in order, so <=8 guarantees the previous chunk landed.
static __device__ __forceinline__ void wait_asynccnt8() {
  asm volatile("s_wait_asynccnt 0x8" ::: "memory");
}

// A-fragment 16x32 bf16 (ISA 7.12.2): lane = m + 16*h holds row m,
// K in [8h,8h+8) for elems 0..7 and [16+8h,16+8h+8) for elems 8..15.
static __device__ __forceinline__ v16bf loadA_f32(const float* src, int ld,
                                                  int row0, int k0, int lane) {
  const int m = lane & 15, h = lane >> 4;
  const float* p = src + (size_t)(row0 + m) * ld + k0 + 8 * h;
  v16bf a;
#pragma unroll
  for (int e = 0; e < 8; ++e) a[e] = (__bf16)p[e];
#pragma unroll
  for (int e = 0; e < 8; ++e) a[8 + e] = (__bf16)p[16 + e];
  return a;
}

static __device__ __forceinline__ v16bf loadA_bf16(const __bf16* src, int ld,
                                                   int row0, int k0, int lane) {
  const int m = lane & 15, h = lane >> 4;
  const __bf16* p = src + (size_t)(row0 + m) * ld + k0 + 8 * h;
  v16bf a;
#pragma unroll
  for (int e = 0; e < 8; ++e) a[e] = p[e];
#pragma unroll
  for (int e = 0; e < 8; ++e) a[8 + e] = p[16 + e];
  return a;
}

// B-fragment 32x16 bf16: lane = n + 16*g holds column n, contraction
// K = k0 + 8*g + e for elems 0..7 and k0 + 16 + 8*g + (e-8) for elems 8..15.
static __device__ __forceinline__ v16bf loadB_f32(const float* src, int ld,
                                                  int n0, int k0, int lane) {
  const int n = lane & 15, kb = (lane >> 4) * 8;
  const float* p = src + (size_t)(n0 + n) * ld + k0 + kb;
  v16bf b;
#pragma unroll
  for (int e = 0; e < 8; ++e) b[e] = (__bf16)p[e];
#pragma unroll
  for (int e = 0; e < 8; ++e) b[8 + e] = (__bf16)p[16 + e];
  return b;
}

static __device__ __forceinline__ v16bf loadB_bf16(const __bf16* src, int ld,
                                                   int n0, int k0, int lane) {
  const int n = lane & 15, kb = (lane >> 4) * 8;
  const __bf16* p = src + (size_t)(n0 + n) * ld + k0 + kb;
  v16bf b;
#pragma unroll
  for (int e = 0; e < 8; ++e) b[e] = p[e];
#pragma unroll
  for (int e = 0; e < 8; ++e) b[8 + e] = p[16 + e];
  return b;
}

// ---------------------------------------------------------------------------
// Kernel 1: complex projection  out = (Xr + iXi) @ (Wr + iWi)^T, cast to bf16.
// X: (4096, 256) f32.  W: (512, 256) f32.  Complex = 4 WMMA chains.
// mode 0: store as (b, h, n, d)            [Q, K]
// mode 1: store transposed (b, h, d, n)    [V, so PV B-frags are contiguous]
// ---------------------------------------------------------------------------
__global__ __launch_bounds__(128) void proj_kernel(
    const float* __restrict__ Xr, const float* __restrict__ Xi,
    const float* __restrict__ Wr, const float* __restrict__ Wi,
    __bf16* __restrict__ outRe, __bf16* __restrict__ outIm, int mode) {
  const int lane = threadIdx.x & 31;
  const int wid  = threadIdx.x >> 5;
  const int id   = blockIdx.x * 4 + wid;           // 8192 tiles total
  const int row0 = (id >> 5) * 16;                 // 256 row tiles
  const int col0 = (id & 31) * 16;                 // 32 col tiles

  v8f arr = v8f_zero(), aii = v8f_zero(), ari = v8f_zero(), air = v8f_zero();
#pragma unroll
  for (int kc = 0; kc < RDIM; kc += 32) {
    v16bf xr = loadA_f32(Xr, RDIM, row0, kc, lane);
    v16bf xi = loadA_f32(Xi, RDIM, row0, kc, lane);
    v16bf wr = loadB_f32(Wr, RDIM, col0, kc, lane);
    v16bf wi = loadB_f32(Wi, RDIM, col0, kc, lane);
    arr = wmma_bf16(xr, wr, arr);
    aii = wmma_bf16(xi, wi, aii);
    ari = wmma_bf16(xr, wi, ari);
    air = wmma_bf16(xi, wr, air);
  }

  const int hh = lane >> 4, cc = lane & 15;
#pragma unroll
  for (int v = 0; v < 8; ++v) {
    const int g = row0 + v + 8 * hh;               // global row
    const int b = g >> 10, n = g & 1023;
    const int c = col0 + cc;                       // global col (h*64+d)
    const int h = c >> 6, d = c & 63;
    const float re = arr[v] - aii[v];
    const float im = ari[v] + air[v];
    size_t idx;
    if (mode == 0)
      idx = (((size_t)(b * NHH + h)) * NSEQ + n) * DHD + d;
    else
      idx = (((size_t)(b * NHH + h)) * DHD + d) * NSEQ + n;
    outRe[idx] = (__bf16)re;
    outIm[idx] = (__bf16)im;
  }
}

// ---------------------------------------------------------------------------
// Kernel 2: fused complex flash attention per (b,h).
// Block = 4 waves = 64 query rows. K/V planes (Kr/Ki/Vr/Vi, 4 KB each) are
// staged into LDS with double-buffered global_load_async_to_lds_b128: each
// iteration issues the NEXT chunk's DMA, waits asynccnt<=8 (previous chunk
// complete, next still in flight), barriers, and computes on the current
// buffer -- the async engine overlaps the whole transfer with WMMA/softmax.
// S = Q conj(K)^T : re = QrKr^T + QiKi^T ; im = QiKr^T - QrKi^T  (WMMA)
// p = softmax(|S|/8, masked)  with online max/sum (shfl width-16 reductions)
// O += p @ V (complex, P staged C-layout -> LDS -> A-layout)
// ---------------------------------------------------------------------------
__global__ __launch_bounds__(128) void attn_kernel(
    const __bf16* __restrict__ Qr, const __bf16* __restrict__ Qi,
    const __bf16* __restrict__ Kr, const __bf16* __restrict__ Ki,
    const __bf16* __restrict__ Vr, const __bf16* __restrict__ Vi,
    const unsigned char* __restrict__ mask,
    __bf16* __restrict__ Ar, __bf16* __restrict__ Ai) {
  // K stage: [buf][plane][key(32)][d(64)]  (row-major, 4 KB/plane)
  // V stage: [buf][plane][d(64)][key(32)]  (row-major, 4 KB/plane)
  __shared__ __align__(16) __bf16 kstage[2][2][32 * 64];
  __shared__ __align__(16) __bf16 vstage[2][2][64 * 32];
  __shared__ float pstage[4][16 * 32];             // per-wave P staging, 8 KB

  const int lane = threadIdx.x & 31;
  const int wid  = threadIdx.x >> 5;
  const int bh   = blockIdx.y;
  const int b    = bh >> 3;
  const int h    = bh & 7;
  const int q0   = (blockIdx.x * 4 + wid) * 16;

  const __bf16* Qrh = Qr + (size_t)bh * NSEQ * DHD;
  const __bf16* Qih = Qi + (size_t)bh * NSEQ * DHD;
  const __bf16* Krh = Kr + (size_t)bh * NSEQ * DHD;
  const __bf16* Kih = Ki + (size_t)bh * NSEQ * DHD;
  const __bf16* Vrh = Vr + (size_t)bh * DHD * NSEQ;   // (d, Nk)
  const __bf16* Vih = Vi + (size_t)bh * DHD * NSEQ;
  const unsigned char* mrow_ptr = mask + (size_t)b * NSEQ;

  // This wave's staging assignment (wave-uniform): one plane per wave.
  const __bf16* gplane = (wid == 0) ? Krh : (wid == 1) ? Kih
                        : (wid == 2) ? Vrh : Vih;
  __bf16* lp0 = (wid == 0) ? kstage[0][0] : (wid == 1) ? kstage[0][1]
               : (wid == 2) ? vstage[0][0] : vstage[0][1];
  __bf16* lp1 = (wid == 0) ? kstage[1][0] : (wid == 1) ? kstage[1][1]
               : (wid == 2) ? vstage[1][0] : vstage[1][1];
  const unsigned lb[2] = {(unsigned)(uintptr_t)lp0,    // LDS = addr[31:0]
                          (unsigned)(uintptr_t)lp1};

  // Issue the 8 async b128 loads that stage this wave's 4 KB plane for a
  // 32-key chunk starting at key kc into the buffer at LDS base ldsb.
  auto issue_chunk = [&](int kc, unsigned ldsb) {
    if (wid < 2) {
      // K plane: keys kc..kc+31 are a contiguous 4 KB block of (key,d) rows.
#pragma unroll
      for (int issue = 0; issue < 8; ++issue) {
        const unsigned unit = issue * 32 + lane;          // 16B units
        async_copy16(ldsb + unit * 16,
                     (unsigned)kc * 128u + unit * 16, gplane);
      }
    } else {
      // V plane: 64 rows (d) of 64 bytes, global row stride NSEQ*2 bytes.
#pragma unroll
      for (int issue = 0; issue < 8; ++issue) {
        const unsigned unit = issue * 32 + lane;
        const unsigned row = unit >> 2, c4 = unit & 3;
        async_copy16(ldsb + row * 64 + c4 * 16,
                     row * (NSEQ * 2u) + (unsigned)kc * 2u + c4 * 16, gplane);
      }
    }
  };

  // Q fragments: resident for the whole sweep (d split 0..31 / 32..63).
  const v16bf qr0 = loadA_bf16(Qrh, DHD, q0, 0, lane);
  const v16bf qr1 = loadA_bf16(Qrh, DHD, q0, 32, lane);
  const v16bf qi0 = loadA_bf16(Qih, DHD, q0, 0, lane);
  const v16bf qi1 = loadA_bf16(Qih, DHD, q0, 32, lane);

  v8f ore[4], oim[4];
#pragma unroll
  for (int dt = 0; dt < 4; ++dt) { ore[dt] = v8f_zero(); oim[dt] = v8f_zero(); }

  float mrun[8], lrun[8];
#pragma unroll
  for (int v = 0; v < 8; ++v) { mrun[v] = -1e30f; lrun[v] = 0.f; }
  int anyacc = 0;

  const int hh = lane >> 4, cc = lane & 15;
  float* myp = &pstage[wid][0];

  // Prologue: stage chunk 0 into buffer 0.
  issue_chunk(0, lb[0]);

  for (int it = 0; it < NSEQ / 32; ++it) {
    const int kc = it * 32;
    const int cur = it & 1;

    // Pipeline: issue next chunk's DMA into the other buffer, then wait only
    // until the CURRENT chunk has landed (<=8 outstanding, in-order).
    if (it + 1 < NSEQ / 32) {                      // uniform branch
      issue_chunk(kc + 32, lb[cur ^ 1]);
      wait_asynccnt8();
    } else {
      wait_asynccnt0();
    }
    __syncthreads();

    const __bf16* kr_s = kstage[cur][0];
    const __bf16* ki_s = kstage[cur][1];
    const __bf16* vr_s = vstage[cur][0];
    const __bf16* vi_s = vstage[cur][1];

    // ---- scores for this chunk (two 16-key tiles) ----
    float sabs[2][8];
    int mk[2];
#pragma unroll
    for (int t = 0; t < 2; ++t) {
      v16bf kr0 = loadB_bf16(kr_s, DHD, t * 16, 0, lane);
      v16bf kr1 = loadB_bf16(kr_s, DHD, t * 16, 32, lane);
      v16bf ki0 = loadB_bf16(ki_s, DHD, t * 16, 0, lane);
      v16bf ki1 = loadB_bf16(ki_s, DHD, t * 16, 32, lane);

      v8f sre = v8f_zero();
      sre = wmma_bf16(qr0, kr0, sre);
      sre = wmma_bf16(qr1, kr1, sre);
      sre = wmma_bf16(qi0, ki0, sre);
      sre = wmma_bf16(qi1, ki1, sre);
      v8f sip = v8f_zero();                        // +Qi Kr^T
      sip = wmma_bf16(qi0, kr0, sip);
      sip = wmma_bf16(qi1, kr1, sip);
      v8f sin = v8f_zero();                        // -Qr Ki^T
      sin = wmma_bf16(qr0, ki0, sin);
      sin = wmma_bf16(qr1, ki1, sin);

      mk[t] = (int)mrow_ptr[kc + t * 16 + cc];
#pragma unroll
      for (int v = 0; v < 8; ++v) {
        const float im = sip[v] - sin[v];
        const float a  = sqrtf(sre[v] * sre[v] + im * im) * 0.125f; // /sqrt(64)
        sabs[t][v] = mk[t] ? a : -1e30f;
      }
    }
    anyacc |= mk[0] | mk[1];

    // ---- online softmax update ----
    float pv0[8], pv1[8];
#pragma unroll
    for (int v = 0; v < 8; ++v) {
      // row lives in one 16-lane half -> width-16 butterfly reductions
      float tm = fmaxf(sabs[0][v], sabs[1][v]);
      tm = fmaxf(tm, __shfl_xor(tm, 1, 16));
      tm = fmaxf(tm, __shfl_xor(tm, 2, 16));
      tm = fmaxf(tm, __shfl_xor(tm, 4, 16));
      tm = fmaxf(tm, __shfl_xor(tm, 8, 16));
      const float mn = fmaxf(mrun[v], tm);
      const float sc = __expf(mrun[v] - mn);
      mrun[v] = mn;
      pv0[v] = __expf(sabs[0][v] - mn);
      pv1[v] = __expf(sabs[1][v] - mn);
      float ps = pv0[v] + pv1[v];
      ps += __shfl_xor(ps, 1, 16);
      ps += __shfl_xor(ps, 2, 16);
      ps += __shfl_xor(ps, 4, 16);
      ps += __shfl_xor(ps, 8, 16);
      lrun[v] = lrun[v] * sc + ps;
#pragma unroll
      for (int dt = 0; dt < 4; ++dt) {
        ore[dt][v] = ore[dt][v] * sc;
        oim[dt][v] = oim[dt][v] * sc;
      }
    }

    // ---- P: C-layout regs -> per-wave LDS -> A-layout fragment ----
#pragma unroll
    for (int v = 0; v < 8; ++v) {
      const int row = v + 8 * hh;
      myp[row * 32 + cc]      = pv0[v];
      myp[row * 32 + 16 + cc] = pv1[v];
    }
    v16bf pa;
    {
      const int m = lane & 15, h2 = lane >> 4;
      const float* pr = myp + m * 32 + 8 * h2;
#pragma unroll
      for (int e = 0; e < 8; ++e) pa[e] = (__bf16)pr[e];
#pragma unroll
      for (int e = 0; e < 8; ++e) pa[8 + e] = (__bf16)pr[16 + e];
    }

    // ---- O += P @ V  (V stage is (d, key): B-frags read two 8-key runs per lane) ----
#pragma unroll
    for (int dt = 0; dt < 4; ++dt) {
      v16bf vr = loadB_bf16(vr_s, 32, dt * 16, 0, lane);
      v16bf vi = loadB_bf16(vi_s, 32, dt * 16, 0, lane);
      ore[dt] = wmma_bf16(pa, vr, ore[dt]);
      oim[dt] = wmma_bf16(pa, vi, oim[dt]);
    }

    __syncthreads();   // all waves done reading buf[cur] before it is refilled
  }

  // Epilogue: normalize by row sum; rows with no valid key output zero.
  const unsigned long long anyv = __ballot(anyacc != 0);
  float inv[8];
#pragma unroll
  for (int v = 0; v < 8; ++v)
    inv[v] = (anyv != 0ULL && lrun[v] > 0.f) ? (1.f / lrun[v]) : 0.f;

#pragma unroll
  for (int dt = 0; dt < 4; ++dt) {
#pragma unroll
    for (int v = 0; v < 8; ++v) {
      const int q = q0 + v + 8 * hh;
      const int d = dt * 16 + cc;
      const size_t o = ((size_t)(b * NSEQ + q)) * CPROJ + h * DHD + d;
      Ar[o] = (__bf16)(ore[dt][v] * inv[v]);
      Ai[o] = (__bf16)(oim[dt][v] * inv[v]);
    }
  }
}

// ---------------------------------------------------------------------------
// Kernel 3: complex output projection  out = (Ar + iAi) @ (WOr + iWOi)^T
// attn: (4096, 512) bf16.  WO: (256, 512) f32.  Output complex64 interleaved.
// ---------------------------------------------------------------------------
__global__ __launch_bounds__(128) void outproj_kernel(
    const __bf16* __restrict__ Ar, const __bf16* __restrict__ Ai,
    const float* __restrict__ WOr, const float* __restrict__ WOi,
    float2* __restrict__ out) {
  const int lane = threadIdx.x & 31;
  const int wid  = threadIdx.x >> 5;
  const int id   = blockIdx.x * 4 + wid;           // 4096 tiles
  const int row0 = (id >> 4) * 16;                 // 256 row tiles
  const int col0 = (id & 15) * 16;                 // 16 col tiles

  v8f arr = v8f_zero(), aii = v8f_zero(), ari = v8f_zero(), air = v8f_zero();
#pragma unroll
  for (int kc = 0; kc < CPROJ; kc += 32) {
    v16bf xr = loadA_bf16(Ar, CPROJ, row0, kc, lane);
    v16bf xi = loadA_bf16(Ai, CPROJ, row0, kc, lane);
    v16bf wr = loadB_f32(WOr, CPROJ, col0, kc, lane);
    v16bf wi = loadB_f32(WOi, CPROJ, col0, kc, lane);
    arr = wmma_bf16(xr, wr, arr);
    aii = wmma_bf16(xi, wi, aii);
    ari = wmma_bf16(xr, wi, ari);
    air = wmma_bf16(xi, wr, air);
  }

  const int hh = lane >> 4, cc = lane & 15;
#pragma unroll
  for (int v = 0; v < 8; ++v) {
    const int g = row0 + v + 8 * hh;
    float2 o;
    o.x = arr[v] - aii[v];
    o.y = ari[v] + air[v];
    out[(size_t)g * RDIM + col0 + cc] = o;
  }
}

// ---------------------------------------------------------------------------
// Host side
// ---------------------------------------------------------------------------
extern "C" void kernel_launch(void* const* d_in, const int* in_sizes, int n_in,
                              void* d_out, int out_size, void* d_ws,
                              size_t ws_size, hipStream_t stream) {
  (void)in_sizes; (void)n_in; (void)out_size; (void)ws_size;

  const float* Qre = (const float*)d_in[0];
  const float* Qim = (const float*)d_in[1];
  const float* Kre = (const float*)d_in[2];
  const float* Kim = (const float*)d_in[3];
  const float* Vre = (const float*)d_in[4];
  const float* Vim = (const float*)d_in[5];
  const float* WQr = (const float*)d_in[6];
  const float* WQi = (const float*)d_in[7];
  const float* WKr = (const float*)d_in[8];
  const float* WKi = (const float*)d_in[9];
  const float* WVr = (const float*)d_in[10];
  const float* WVi = (const float*)d_in[11];
  const float* WOr = (const float*)d_in[12];
  const float* WOi = (const float*)d_in[13];
  const unsigned char* mask = (const unsigned char*)d_in[14];

  // Workspace: 8 bf16 planes of (B*NH*1024*64) = 4 MB each -> 32 MB total.
  const size_t PE = (size_t)NB * NHH * NSEQ * DHD;   // 2,097,152 elements
  char* ws = (char*)d_ws;
  __bf16* Qpr = (__bf16*)(ws + 0 * PE * 2);
  __bf16* Qpi = (__bf16*)(ws + 1 * PE * 2);
  __bf16* Kpr = (__bf16*)(ws + 2 * PE * 2);
  __bf16* Kpi = (__bf16*)(ws + 3 * PE * 2);
  __bf16* Vtr = (__bf16*)(ws + 4 * PE * 2);          // transposed (b,h,d,n)
  __bf16* Vti = (__bf16*)(ws + 5 * PE * 2);
  __bf16* Aar = (__bf16*)(ws + 6 * PE * 2);          // attn (b,q,h*64+d)
  __bf16* Aai = (__bf16*)(ws + 7 * PE * 2);

  // 8192 tiles / 4 waves per block
  proj_kernel<<<2048, 128, 0, stream>>>(Qre, Qim, WQr, WQi, Qpr, Qpi, 0);
  proj_kernel<<<2048, 128, 0, stream>>>(Kre, Kim, WKr, WKi, Kpr, Kpi, 0);
  proj_kernel<<<2048, 128, 0, stream>>>(Vre, Vim, WVr, WVi, Vtr, Vti, 1);

  // grid: (Nq / (16 rows * 4 waves), B*NH)
  attn_kernel<<<dim3(16, 32), 128, 0, stream>>>(Qpr, Qpi, Kpr, Kpi, Vtr, Vti,
                                                mask, Aar, Aai);

  // 4096 tiles / 4 waves per block
  outproj_kernel<<<1024, 128, 0, stream>>>(Aar, Aai, WOr, WOi, (float2*)d_out);
}
